// Wisard_43233140801687
// MI455X (gfx1250) — matrix operand order repaired and checked by reading, not compiled, not run on hardware
//
#include <hip/hip_runtime.h>

// WiSARD inference for MI455X (gfx1250, wave32).
//
// out[b,c] = sum_n ram[c, n, addr(b,c,n)], addr = big-endian pack of 16
// permuted sample bits. Gather-bound: ~2.6M random 4B gathers into a 168MB
// table (fits the 192MB L2); HBM floor ~15-30us at 23.3 TB/s. Address
// packing is tensorized with v_wmma_f32_16x16x32_f16 (bits x power-of-two
// weights, exact in f32 accumulation).
//
// Structure: one block = one sample b (320 threads = 10 waves), wave w =
// class c. The sample row is ballot-packed into LDS once per block and
// shared by all 10 classes. Each wave: 2 WMMAs -> 64 exact addresses ->
// 64 ram gathers -> ds_swizzle butterfly reduction -> one store.

typedef __attribute__((ext_vector_type(16))) _Float16 v16h;
typedef __attribute__((ext_vector_type(8)))  float    v8f;
typedef __attribute__((ext_vector_type(8)))  unsigned u32x8;

#define BATCH   4096
#define ENTRY   1024
#define NCLS    10
#define NNEU    64
#define TUP     16
#define WORDS   (ENTRY / 32)     // 32 packed words per sample
#define BLOCK   (NCLS * 32)      // 320 threads = 10 waves (one class each)

// ---------------------------------------------------------------------------
// Kernel 1: reorder tuple_mapping into the exact gather order the WMMA
// A-operand needs, so the hot loop reads 64 contiguous bytes per lane.
//
// WMMA A (16-bit, 16x32, wave32) layout:
//   lane L (row M = L&15), element e of v16h:
//     lanes 0-15 : e=0..7 -> K=e   ; e=8..15 -> K=e+8
//     lanes 16-31: e=0..7 -> K=e+8 ; e=8..15 -> K=e+16
//   => tuple group g = L>>4, bit t = e (B uses the same K order as A)
// Row m carries neurons (2m) and (2m+1) of the current 32-neuron half.
// ---------------------------------------------------------------------------
__global__ void wisard_tm_reorder(const int* __restrict__ tm,
                                  int* __restrict__ tmr) {
  int idx = blockIdx.x * blockDim.x + threadIdx.x;
  if (idx >= NCLS * 2 * 32 * 16) return;
  int e  = idx & 15;
  int L  = (idx >> 4) & 31;
  int r  = (idx >> 9) & 1;          // which 32-neuron half
  int c  = idx >> 10;
  int m  = L & 15;
  int hi = L >> 4;
  int g  = hi;
  int t  = e;                       // bit index within tuple (big-endian pack)
  int n  = 32 * r + 2 * m + g;
  tmr[idx] = tm[c * ENTRY + n * TUP + t];
}

// 8-way per-lane select from the WMMA accumulator (v_cndmask tree).
__device__ __forceinline__ float sel8(v8f c, int r) {
  float v01 = (r & 1) ? c[1] : c[0];
  float v23 = (r & 1) ? c[3] : c[2];
  float v45 = (r & 1) ? c[5] : c[4];
  float v67 = (r & 1) ? c[7] : c[6];
  float v03 = (r & 2) ? v23 : v01;
  float v47 = (r & 2) ? v67 : v45;
  return (r & 4) ? v47 : v03;
}

// 32-lane butterfly sum via ds_swizzle group-of-32 XOR patterns (no bounds
// checks, unlike __shfl_xor). offset = (xor_mask<<10) | 0x1F.
__device__ __forceinline__ float wave_reduce_add(float v) {
#if __has_builtin(__builtin_amdgcn_ds_swizzle)
  v += __builtin_bit_cast(float, __builtin_amdgcn_ds_swizzle(__builtin_bit_cast(int, v), 0x041F));
  v += __builtin_bit_cast(float, __builtin_amdgcn_ds_swizzle(__builtin_bit_cast(int, v), 0x081F));
  v += __builtin_bit_cast(float, __builtin_amdgcn_ds_swizzle(__builtin_bit_cast(int, v), 0x101F));
  v += __builtin_bit_cast(float, __builtin_amdgcn_ds_swizzle(__builtin_bit_cast(int, v), 0x201F));
  v += __builtin_bit_cast(float, __builtin_amdgcn_ds_swizzle(__builtin_bit_cast(int, v), 0x401F));
#else
  #pragma unroll
  for (int off = 16; off > 0; off >>= 1) v += __shfl_xor(v, off, 32);
#endif
  return v;
}

// ---------------------------------------------------------------------------
__global__ __launch_bounds__(BLOCK) void wisard_main(
    const int*   __restrict__ samples,   // [B, E] 0/1 int32
    const int*   __restrict__ tmr,       // reordered tuple_mapping (ws)
    const float* __restrict__ ram,       // [C, N, 65536]
    float*       __restrict__ out)       // [B, C]
{
  // Packed sample row, duplicated: words 0..31 (banks 0-31) for lanes 0-15,
  // words 32..63 (banks 32-63) for lanes 16-31 -> halves the bank-conflict
  // collision domain on the random per-bit ds reads.
  __shared__ unsigned s_words[2 * WORDS];     // 256 B
  const int lane = threadIdx.x & 31;
  const int wave = threadIdx.x >> 5;          // class c
  const int b    = blockIdx.x;
  const int c    = wave;

  // --- stage packed bits of sample row b once per block (cooperative) ---
#if __has_builtin(__builtin_amdgcn_ballot_w32)
  for (int it = wave; it < WORDS; it += NCLS) {        // waves 0,1 do 4; rest 3
    int sbit = samples[b * ENTRY + it * 32 + lane] & 1;      // coalesced 128B
    unsigned m = __builtin_amdgcn_ballot_w32(sbit != 0);     // word it, LSB=lane0
    if (lane == 0) { s_words[it] = m; s_words[WORDS + it] = m; }
  }
#else
  for (int w = threadIdx.x; w < WORDS; w += BLOCK) {
    unsigned m = 0;
    for (int j = 0; j < 32; ++j)
      m |= (unsigned)(samples[b * ENTRY + w * 32 + j] & 1) << j;
    s_words[w] = m; s_words[WORDS + w] = m;
  }
#endif
  __syncthreads();

  // --- B operand, assembled as raw f16 bit patterns (no float converts) ---
  // f16 2^k = (k+15)<<10. Element e wants 2^(15-e): pair i packs exponents
  // (30-2i, 29-2i). Active only on lanes where (lane>=16)==(lane&1): B
  // with K relabelled contiguously (16-bit 32x16): lanes 0-15 N=lane,K=e; lanes 16-31 N=lane-16,
  // K=e+16; B[k][n] = ((k>>4)==(n&1)) ? 2^(15-(k&15)) : 0, so column parity
  // selects the tuple group: C[m][n] = addr(row m, group n&1).
  const bool act = ((lane >> 4) & 1) == (lane & 1);
  u32x8 bu;
  #pragma unroll
  for (int i = 0; i < 8; ++i) {
    unsigned pairbits = ((unsigned)(30 - 2 * i) << 10)
                      | ((unsigned)(29 - 2 * i) << 26);
    bu[i] = act ? pairbits : 0u;
  }
  const v16h bm = __builtin_bit_cast(v16h, bu);

  const unsigned* wrow = &s_words[(lane & 16) << 1];   // replica steer

  float sum = 0.0f;
  #pragma unroll
  for (int r = 0; r < 2; ++r) {               // two 32-neuron halves
    // 64B contiguous per lane, lane-adjacent -> coalesced b128 loads
    const int4* tp = (const int4*)&tmr[(((c * 2 + r) * 32) + lane) * 16];
    int4 q0 = tp[0], q1 = tp[1], q2 = tp[2], q3 = tp[3];
    const int pos[16] = { q0.x, q0.y, q0.z, q0.w, q1.x, q1.y, q1.z, q1.w,
                          q2.x, q2.y, q2.z, q2.w, q3.x, q3.y, q3.z, q3.w };
    // A operand: f16 1.0 = 0x3C00, so pair = (b0 | b1<<16) * 0x3C00.
    u32x8 au;
    #pragma unroll
    for (int i = 0; i < 8; ++i) {
      int p0 = pos[2 * i], p1 = pos[2 * i + 1];
      unsigned b0 = (wrow[p0 >> 5] >> (p0 & 31)) & 1u;   // ds_load_b32
      unsigned b1 = (wrow[p1 >> 5] >> (p1 & 31)) & 1u;
      au[i] = (b0 | (b1 << 16)) * 0x3C00u;
    }
    v16h a = __builtin_bit_cast(v16h, au);
    v8f acc = {};
    // (neg_a, A, neg_b, B, c_mod, C, reuse_a, reuse_b)
    acc = __builtin_amdgcn_wmma_f32_16x16x32_f16(
        false, a, false, bm, (short)0, acc, false, false);
    // lane L -> neuron 32*r + L, address at accumulator slot (L&15)>>1
    int addr = (int)sel8(acc, (lane & 15) >> 1);   // exact integer in [0,65536)
    int n = 32 * r + lane;
    sum += ram[((c * NNEU + n) << 16) + addr];     // random gather, L2-resident
  }

  sum = wave_reduce_add(sum);                      // 64-neuron total for (b,c)
  if (lane == 0) out[b * NCLS + c] = sum;
}

// ---------------------------------------------------------------------------
extern "C" void kernel_launch(void* const* d_in, const int* in_sizes, int n_in,
                              void* d_out, int out_size, void* d_ws, size_t ws_size,
                              hipStream_t stream) {
  const int*   samples = (const int*)d_in[0];    // [4096,1024] int32
  const int*   tm      = (const int*)d_in[1];    // [10,1024] int32 (JAX x32)
  const float* ram     = (const float*)d_in[2];  // [10,64,65536] float32
  float*       out     = (float*)d_out;          // [4096,10] float32
  int*         tmr     = (int*)d_ws;             // 40 KB scratch

  const int n1 = NCLS * 2 * 32 * 16;             // 10240 reordered indices
  wisard_tm_reorder<<<(n1 + 255) / 256, 256, 0, stream>>>(tm, tmr);
  wisard_main<<<BATCH, BLOCK, 0, stream>>>(samples, tmr, ram, out);
}
